// CrossAtt_23313082483415
// MI455X (gfx1250) — compile-verified, matrix operand order checked by reading, not run on hardware
//
#include <hip/hip_runtime.h>
#include <cstdint>
#include <cstddef>

typedef __attribute__((ext_vector_type(16))) __bf16 v16bf;
typedef __attribute__((ext_vector_type(8)))  __bf16 v8bf;
typedef __attribute__((ext_vector_type(8)))  float  v8f;

#define HW   4096
#define CCH  256
#define BB   4
#define NPB  (BB * CCH * HW)   // elements per output tensor

__device__ __forceinline__ __bf16 f2bf(float f) {
  uint32_t u = __builtin_bit_cast(uint32_t, f);
  u += 0x7FFFu + ((u >> 16) & 1u);            // round-to-nearest-even
  unsigned short h = (unsigned short)(u >> 16);
  return __builtin_bit_cast(__bf16, h);
}

// Per-lane 16-bit fragment (ISA 7.12.2): lane's 16 elements are K = koff+0..7
// and K = koff+16..23 -> two contiguous 16B runs when the K axis is the fast
// LDS axis.  base must point at element (row, koff) and be 16B aligned.
__device__ __forceinline__ v16bf load_frag(const __bf16* base) {
  const uint4* p = (const uint4*)base;
  uint4 a = p[0];   // K = koff + 0..7
  uint4 b = p[2];   // K = koff + 16..23
  v8bf lo = __builtin_bit_cast(v8bf, a);
  v8bf hi = __builtin_bit_cast(v8bf, b);
  return __builtin_shufflevector(lo, hi, 0, 1, 2, 3, 4, 5, 6, 7,
                                 8, 9, 10, 11, 12, 13, 14, 15);
}

__device__ __forceinline__ float redmax16(float v) {
#pragma unroll
  for (int m = 1; m < 16; m <<= 1) v = fmaxf(v, __shfl_xor(v, m, 32));
  return v;
}
__device__ __forceinline__ float redsum16(float v) {
#pragma unroll
  for (int m = 1; m < 16; m <<= 1) v += __shfl_xor(v, m, 32);
  return v;
}

__device__ __forceinline__ v8f wmma_bf16(const v16bf& a, const v16bf& b, v8f c) {
  return __builtin_amdgcn_wmma_f32_16x16x32_bf16(false, a, false, b, (short)0, c,
                                                 false, false);
}

// ---------------------------------------------------------------------------
// K0: fold BN into 3x3 weights -> bf16, layout wB[p=ky*3+kx][oc][ic]
// (ic fastest so conv B-fragments are K-contiguous).
// ---------------------------------------------------------------------------
__global__ __launch_bounds__(256) void prep_kernel(
    const float* __restrict__ w_cat, const float* __restrict__ bn_scale,
    const float* __restrict__ bn_bias, const float* __restrict__ bn_mean,
    const float* __restrict__ bn_var, __bf16* __restrict__ wB,
    float* __restrict__ biasp) {
  int idx = blockIdx.x * 256 + threadIdx.x;
  if (idx < 256) {
    float inv = bn_scale[idx] * rsqrtf(bn_var[idx] + 1e-5f);
    biasp[idx] = bn_bias[idx] - bn_mean[idx] * inv;
  }
  if (idx >= 9 * 256 * 512) return;
  int ic = idx & 511;
  int oc = (idx >> 9) & 255;
  int p  = idx >> 17;
  float inv = bn_scale[oc] * rsqrtf(bn_var[oc] + 1e-5f);
  wB[idx] = f2bf(w_cat[(size_t)oc * 4608 + ic * 9 + p] * inv);
}

// ---------------------------------------------------------------------------
// K1: 1x1 conv projections -> bf16.
//   qb[b][n][64] (q1 || q2), k{1,2}b[b][n][64]                (row = K-contig)
//   v{1,2}b[b][nblk=n/64][ch 256][key n%64]  (block-transposed for flash B)
// ---------------------------------------------------------------------------
__global__ __launch_bounds__(256) void qkv_kernel(
    const float* __restrict__ in1, const float* __restrict__ in2,
    const float* __restrict__ wq1, const float* __restrict__ bq1,
    const float* __restrict__ wk1, const float* __restrict__ bk1,
    const float* __restrict__ wv1, const float* __restrict__ bv1,
    const float* __restrict__ wq2, const float* __restrict__ bq2,
    const float* __restrict__ wk2, const float* __restrict__ bk2,
    const float* __restrict__ wv2, const float* __restrict__ bv2,
    __bf16* __restrict__ qb, __bf16* __restrict__ k1b, __bf16* __restrict__ k2b,
    __bf16* __restrict__ v1b, __bf16* __restrict__ v2b) {
  int blk = blockIdx.x;
  int b   = blk >> 8;
  int n0  = (blk & 255) * 16;
  int tid = threadIdx.x;

  __shared__ float x1s[256 * 16];
  __shared__ float x2s[256 * 16];
  for (int i = tid; i < 256 * 16; i += 256) {
    int c = i >> 4, p = i & 15;
    size_t g = ((size_t)b * CCH + c) * HW + n0 + p;
    x1s[i] = in1[g];
    x2s[i] = in2[g];
  }
  __syncthreads();

  for (int j = tid; j < 704 * 16; j += 256) {
    int p = j & 15;
    int o = j >> 4;  // 0..703
    int n = n0 + p;
    const float* w;
    const float* xs;
    float bv;
    __bf16* dst;
    size_t di;
    size_t nrow = (size_t)b * HW + n;
    size_t vrow = ((size_t)b * (HW / 64) + (n >> 6)) * 256;  // v block base /ch
    if (o < 32) {
      w = wq1 + o * 256; bv = bq1[o]; xs = x1s; dst = qb; di = nrow * 64 + o;
    } else if (o < 64) {
      int oo = o - 32;
      w = wq2 + oo * 256; bv = bq2[oo]; xs = x2s; dst = qb; di = nrow * 64 + o;
    } else if (o < 128) {
      int oo = o - 64;
      w = wk1 + oo * 256; bv = bk1[oo]; xs = x1s; dst = k1b; di = nrow * 64 + oo;
    } else if (o < 192) {
      int oo = o - 128;
      w = wk2 + oo * 256; bv = bk2[oo]; xs = x2s; dst = k2b; di = nrow * 64 + oo;
    } else if (o < 448) {
      int oo = o - 192;
      w = wv1 + oo * 256; bv = bv1[oo]; xs = x1s; dst = v1b;
      di = (vrow + oo) * 64 + (n & 63);
    } else {
      int oo = o - 448;
      w = wv2 + oo * 256; bv = bv2[oo]; xs = x2s; dst = v2b;
      di = (vrow + oo) * 64 + (n & 63);
    }
    float s = bv;
#pragma unroll 8
    for (int c = 0; c < 256; ++c) s += w[c] * xs[c * 16 + p];
    dst[di] = f2bf(s);
  }
}

// ---------------------------------------------------------------------------
// K2: flash attention, one branch per blockIdx.z.
// Block = 64 query rows, 256 threads = 8 waves (wave32).
// wave w: r = w&3 -> query rows 16r..16r+15 ; h = w>>2 -> S cols 32h..+31,
//                                                        O chans 128h..+127.
// ---------------------------------------------------------------------------
__global__ __launch_bounds__(256) void attn_kernel(
    const __bf16* __restrict__ qb, const __bf16* __restrict__ k1b,
    const __bf16* __restrict__ k2b, const __bf16* __restrict__ v1b,
    const __bf16* __restrict__ v2b, const float* __restrict__ in1,
    const float* __restrict__ in2, const float* __restrict__ gam,
    float* __restrict__ dout) {
  const int n0 = blockIdx.x * 64;
  const int b  = blockIdx.y;
  const int br = blockIdx.z;
  const __bf16* kptr = (br ? k2b : k1b) + (size_t)b * HW * 64;
  const __bf16* vptr = (br ? v2b : v1b) + (size_t)b * HW * 256;  // [nb][256][64]
  const float*  xin  = (br ? in2 : in1) + (size_t)b * CCH * HW;
  float* outp = dout + (size_t)(1 + br) * NPB + (size_t)b * CCH * HW;
  const __bf16* qptr = qb + ((size_t)b * HW + n0) * 64;

  __shared__ __align__(16) __bf16 Qs[64 * 64];    // [qrow][K=d]
  __shared__ __align__(16) __bf16 Ks[64 * 64];    // [kcol][K=d]
  __shared__ __align__(16) __bf16 Pld[64 * 64];   // [qrow][K=key]
  __shared__ __align__(16) __bf16 Vst[256 * 64];  // [ch][K=key]
  __shared__ float m_s[64], l_s[64], alpha_s[64];
  __shared__ float rmax[64][2], rsum[64][2];

  const int tid  = threadIdx.x;
  const int lane = tid & 31, wid = tid >> 5;
  const int r = wid & 3, h = wid >> 2;
  const int rl = lane & 15;
  const int koff = (lane < 16) ? 0 : 8;
  const int rowhi = 8 * (lane >> 4);

  {  // load Q tile (contiguous 8KB)
    const uint4* s = (const uint4*)qptr;
    uint4* d = (uint4*)Qs;
    for (int i = tid; i < 512; i += 256) d[i] = s[i];
  }
  if (tid < 64) { m_s[tid] = -1e30f; l_s[tid] = 0.f; }
  __syncthreads();

  v16bf aQ0 = load_frag(&Qs[(16 * r + rl) * 64 + koff]);
  v16bf aQ1 = load_frag(&Qs[(16 * r + rl) * 64 + 32 + koff]);
  v8f acc[8] = {};

  for (int kb0 = 0; kb0 < HW; kb0 += 64) {
    __syncthreads();  // previous iteration done with Ks/Vst/Pld
    {
      const uint4* s = (const uint4*)(kptr + (size_t)kb0 * 64);
      uint4* d = (uint4*)Ks;
      for (int i = tid; i < 512; i += 256) d[i] = s[i];
    }
    {
      const uint4* s = (const uint4*)(vptr + (size_t)kb0 * 256);  // contiguous
      uint4* d = (uint4*)Vst;
      for (int i = tid; i < 2048; i += 256) d[i] = s[i];
    }
    if (kb0 + 64 < HW) {  // hint next tiles -> global_prefetch_b8
      __builtin_prefetch(kptr + (size_t)(kb0 + 64) * 64 + tid * 16, 0, 0);
      __builtin_prefetch(vptr + (size_t)(kb0 + 64) * 256 + tid * 64, 0, 0);
    }
    __syncthreads();

    // ---- S = Q * K^T : all 4 B-fragments issued before the WMMAs ----
    v8f st0 = {}, st1 = {};
    {
      v16bf bK00 = load_frag(&Ks[(32 * h + rl) * 64 + koff]);
      v16bf bK01 = load_frag(&Ks[(32 * h + rl) * 64 + 32 + koff]);
      v16bf bK10 = load_frag(&Ks[(32 * h + 16 + rl) * 64 + koff]);
      v16bf bK11 = load_frag(&Ks[(32 * h + 16 + rl) * 64 + 32 + koff]);
      st0 = wmma_bf16(aQ0, bK00, st0);
      st0 = wmma_bf16(aQ1, bK01, st0);
      st1 = wmma_bf16(aQ0, bK10, st1);
      st1 = wmma_bf16(aQ1, bK11, st1);
    }
    // ---- per-row partial max over this wave's 32 cols ----
#pragma unroll
    for (int i = 0; i < 8; ++i) {
      float v = fmaxf(redmax16(st0[i]), redmax16(st1[i]));
      if (rl == 0) rmax[16 * r + i + rowhi][h] = v;
    }
    __syncthreads();
    if (tid < 64) {
      float mo = m_s[tid];
      float mn = fmaxf(mo, fmaxf(rmax[tid][0], rmax[tid][1]));
      alpha_s[tid] = __expf(mo - mn);
      m_s[tid] = mn;
    }
    __syncthreads();
    // ---- P = exp(S - m), row partial sums, rescale O ----
#pragma unroll
    for (int i = 0; i < 8; ++i) {
      int row = 16 * r + i + rowhi;
      float m = m_s[row];
      float p0 = __expf(st0[i] - m);
      float p1 = __expf(st1[i] - m);
      Pld[row * 64 + 32 * h + rl]      = f2bf(p0);
      Pld[row * 64 + 32 * h + 16 + rl] = f2bf(p1);
      float rs = redsum16(p0) + redsum16(p1);
      if (rl == 0) rsum[row][h] = rs;
      float a = alpha_s[row];
#pragma unroll
      for (int t = 0; t < 8; ++t) acc[t][i] *= a;
    }
    __syncthreads();
    if (tid < 64) l_s[tid] = l_s[tid] * alpha_s[tid] + rsum[tid][0] + rsum[tid][1];

    // ---- O += P * V : prefetch next tile's V-fragments one step ahead ----
    v16bf aP0 = load_frag(&Pld[(16 * r + rl) * 64 + koff]);
    v16bf aP1 = load_frag(&Pld[(16 * r + rl) * 64 + 32 + koff]);
    v16bf bVa = load_frag(&Vst[(128 * h + rl) * 64 + koff]);
    v16bf bVb = load_frag(&Vst[(128 * h + rl) * 64 + 32 + koff]);
#pragma unroll
    for (int t = 0; t < 8; ++t) {
      v16bf nVa = bVa, nVb = bVb;
      if (t < 7) {
        int cb = 128 * h + 16 * (t + 1);
        nVa = load_frag(&Vst[(cb + rl) * 64 + koff]);
        nVb = load_frag(&Vst[(cb + rl) * 64 + 32 + koff]);
      }
      acc[t] = wmma_bf16(aP0, bVa, acc[t]);
      acc[t] = wmma_bf16(aP1, bVb, acc[t]);
      bVa = nVa;
      bVb = nVb;
    }
  }
  __syncthreads();

  const float g = gam[0];
  float invl[8];
#pragma unroll
  for (int i = 0; i < 8; ++i) invl[i] = g / l_s[16 * r + i + rowhi];
#pragma unroll
  for (int t = 0; t < 8; ++t) {
    int ch = 128 * h + 16 * t + rl;
#pragma unroll
    for (int i = 0; i < 8; ++i) {
      int row = 16 * r + i + rowhi;
      float val = acc[t][i] * invl[i] + xin[(size_t)ch * HW + n0 + row];
      outp[(size_t)ch * HW + n0 + row] = val;
    }
  }
}

// ---------------------------------------------------------------------------
// K3: implicit-GEMM 3x3 conv over cat(out1,out2) + fused BN bias + ReLU.
// Block = (batch, image row y, 64 output channels). 8 waves:
//  r = w&3 -> pixels 16r..+15 of the row ; h = w>>2 -> oc 32h + {0,16}.
// K loop: 16 chunks of 32 input channels x 9 taps, one WMMA K-step each,
// with one-tap-ahead fragment prefetch.
// ---------------------------------------------------------------------------
__global__ __launch_bounds__(256) void conv_kernel(
    const float* __restrict__ cat1, const float* __restrict__ cat2,
    const __bf16* __restrict__ wB, const float* __restrict__ biasp,
    float* __restrict__ feat) {
  const int y  = blockIdx.x;
  const int og = blockIdx.y;
  const int b  = blockIdx.z;
  const int tid = threadIdx.x;
  const int lane = tid & 31, wid = tid >> 5;
  const int r = wid & 3, h = wid >> 2;
  const int rl = lane & 15;
  const int koff = (lane < 16) ? 0 : 8;
  const int x = 16 * r + rl;

  __shared__ __align__(16) __bf16 slab[3 * 66 * 32];  // [ky][x+1][ic32]
  __shared__ __align__(16) __bf16 wBs[9 * 64 * 32];   // [tap][oc64][ic32]

  v8f acc[2] = {};
  for (int cc = 0; cc < 16; ++cc) {
    __syncthreads();
    const float* src = (cc < 8 ? cat1 : cat2) + (size_t)b * CCH * HW +
                       (size_t)((cc & 7) * 32) * HW;
    for (int i = tid; i < 3 * 66 * 32; i += 256) {
      int j = i / 198, rem = i % 198;   // j = ic within chunk
      int ky = rem / 66, xx = rem % 66 - 1;
      int yy = y + ky - 1;
      float v = (yy >= 0 && yy < 64 && xx >= 0 && xx < 64)
                    ? src[(size_t)j * HW + yy * 64 + xx] : 0.f;
      slab[(ky * 66 + xx + 1) * 32 + j] = f2bf(v);
    }
    {  // weights: 2304 x uint4 (8 bf16), K-contiguous on both sides
      const uint4* sw = (const uint4*)(wB + (size_t)cc * 32);
      uint4* dw = (uint4*)wBs;
      for (int i = tid; i < 2304; i += 256) {
        int p = i >> 8, rem = i & 255;
        int ocl = rem >> 2, q = rem & 3;
        dw[i] = sw[(((size_t)p * 256 + og * 64 + ocl) * 512) / 8 + q];
      }
    }
    __syncthreads();

    v16bf aA  = load_frag(&slab[(0 * 66 + x + 0) * 32 + koff]);   // p = 0
    v16bf bW0 = load_frag(&wBs[(0 * 64 + 32 * h + rl) * 32 + koff]);
    v16bf bW1 = load_frag(&wBs[(0 * 64 + 32 * h + 16 + rl) * 32 + koff]);
#pragma unroll
    for (int p = 0; p < 9; ++p) {
      v16bf nA = aA, nW0 = bW0, nW1 = bW1;
      if (p < 8) {
        int pn = p + 1;
        int ky = pn / 3, kx = pn % 3;
        nA  = load_frag(&slab[(ky * 66 + x + kx) * 32 + koff]);
        nW0 = load_frag(&wBs[(pn * 64 + 32 * h + rl) * 32 + koff]);
        nW1 = load_frag(&wBs[(pn * 64 + 32 * h + 16 + rl) * 32 + koff]);
      }
      acc[0] = wmma_bf16(aA, bW0, acc[0]);
      acc[1] = wmma_bf16(aA, bW1, acc[1]);
      aA = nA;
      bW0 = nW0;
      bW1 = nW1;
    }
  }

#pragma unroll
  for (int t = 0; t < 2; ++t) {
    int oc = og * 64 + 32 * h + 16 * t + rl;
    float bia = biasp[oc];
#pragma unroll
    for (int i = 0; i < 8; ++i) {
      int xo = 16 * r + i + 8 * (lane >> 4);
      float v = fmaxf(acc[t][i] + bia, 0.f);
      feat[((size_t)b * CCH + oc) * HW + y * 64 + xo] = v;
    }
  }
}

// ---------------------------------------------------------------------------
extern "C" void kernel_launch(void* const* d_in, const int* in_sizes, int n_in,
                              void* d_out, int out_size, void* d_ws, size_t ws_size,
                              hipStream_t stream) {
  const float* in1 = (const float*)d_in[0];
  const float* in2 = (const float*)d_in[1];
  const float* wq1 = (const float*)d_in[2];
  const float* bq1 = (const float*)d_in[3];
  const float* wk1 = (const float*)d_in[4];
  const float* bk1 = (const float*)d_in[5];
  const float* wv1 = (const float*)d_in[6];
  const float* bv1 = (const float*)d_in[7];
  const float* wq2 = (const float*)d_in[8];
  const float* bq2 = (const float*)d_in[9];
  const float* wk2 = (const float*)d_in[10];
  const float* bk2 = (const float*)d_in[11];
  const float* wv2 = (const float*)d_in[12];
  const float* bv2 = (const float*)d_in[13];
  const float* gam = (const float*)d_in[14];
  const float* w_cat = (const float*)d_in[15];
  const float* bn_scale = (const float*)d_in[16];
  const float* bn_bias = (const float*)d_in[17];
  const float* bn_mean = (const float*)d_in[18];
  const float* bn_var = (const float*)d_in[19];

  char* ws = (char*)d_ws;
  __bf16* qb  = (__bf16*)(ws);
  __bf16* k1b = (__bf16*)(ws + (size_t)2  * (1 << 20));
  __bf16* k2b = (__bf16*)(ws + (size_t)4  * (1 << 20));
  __bf16* v1b = (__bf16*)(ws + (size_t)6  * (1 << 20));
  __bf16* v2b = (__bf16*)(ws + (size_t)14 * (1 << 20));
  __bf16* wB  = (__bf16*)(ws + (size_t)22 * (1 << 20));
  float* biasp = (float*)(ws + (size_t)25 * (1 << 20));

  float* dout = (float*)d_out;
  float* out1 = dout + (size_t)NPB;
  float* out2 = dout + (size_t)2 * NPB;

  prep_kernel<<<dim3(4608), 256, 0, stream>>>(w_cat, bn_scale, bn_bias, bn_mean,
                                              bn_var, wB, biasp);
  qkv_kernel<<<dim3(BB * 256), 256, 0, stream>>>(
      in1, in2, wq1, bq1, wk1, bk1, wv1, bv1, wq2, bq2, wk2, bk2, wv2, bv2,
      qb, k1b, k2b, v1b, v2b);
  attn_kernel<<<dim3(HW / 64, BB, 2), 256, 0, stream>>>(
      qb, k1b, k2b, v1b, v2b, in1, in2, gam, dout);
  conv_kernel<<<dim3(64, 4, BB), 256, 0, stream>>>(out1, out2, wB, biasp, dout);
}
